// FocusedLinearSelfAttention_71038759075956
// MI455X (gfx1250) — matrix operand order repaired and checked by reading, not compiled, not run on hardware
//
#include <hip/hip_runtime.h>
#include <hip/hip_bf16.h>

typedef __attribute__((ext_vector_type(16))) _Float16     v16h;
typedef __attribute__((ext_vector_type(8)))  float        v8f;
typedef __attribute__((ext_vector_type(4)))  unsigned int v4u;
typedef __attribute__((ext_vector_type(8)))  int          v8i;
typedef __attribute__((ext_vector_type(4)))  int          v4i;

#define EPSF 1e-6f

#ifndef __has_builtin
#define __has_builtin(x) 0
#endif
#if __has_builtin(__builtin_amdgcn_tensor_load_to_lds) && \
    __has_builtin(__builtin_amdgcn_s_wait_tensorcnt)
#define USE_TDM 1
#else
#define USE_TDM 0
#endif

// ---------------- WMMA fragment index helpers (wave32, 16-bit operands) ----
// A 16x32: lanes 0-15 -> K 0..7,16..23 ; lanes 16-31 -> K 8..15,24..31
__device__ __forceinline__ int a_kidx(int e, int lane) {
    int v = e >> 1, h = e & 1;
    int k = ((v < 4) ? (2 * v) : (16 + 2 * (v - 4))) + h;
    return k + ((lane < 16) ? 0 : 8);
}
// B 32x16: lanes 0-15 -> K 0..7,16..23 ; lanes 16-31 -> K 8..15,24..31
__device__ __forceinline__ int b_kidx(int e, int lane) {
    return ((e < 8) ? e : (e + 8)) + ((lane < 16) ? 0 : 8);
}

__device__ __forceinline__ v16h load_A_frag(const _Float16* s, int ld, int lane) {
    int row = lane & 15;
    v16h a;
#pragma unroll
    for (int e = 0; e < 16; ++e) a[e] = s[row * ld + a_kidx(e, lane)];
    return a;
}
__device__ __forceinline__ v16h load_B_frag_kn(const _Float16* s, int ld, int lane, int coloff) {
    int col = (lane & 15) + coloff;
    v16h b;
#pragma unroll
    for (int e = 0; e < 16; ++e) b[e] = s[b_kidx(e, lane) * ld + col];
    return b;
}
__device__ __forceinline__ v16h load_B_frag_nk(const _Float16* s, int ld, int lane, int coloff) {
    int col = (lane & 15) + coloff;
    v16h b;
#pragma unroll
    for (int e = 0; e < 16; ++e) b[e] = s[col * ld + b_kidx(e, lane)];
    return b;
}

// ---------------- Tensor Data Mover: 2-D f16 tile -> LDS -------------------
#if USE_TDM
// Loads `rows` x `cols` f16 tile (row stride = stride elems) from gptr into
// LDS at byte offset lds_off, packed contiguously [rows][cols].
__device__ __forceinline__ void tdm_load_tile_f16(unsigned lds_off,
                                                  const _Float16* gptr,
                                                  unsigned rows, unsigned cols,
                                                  unsigned stride) {
    unsigned long long ga = (unsigned long long)(__SIZE_TYPE__)gptr;
    // D# group 0: count=1 | lds_addr | global_addr(57b) | type=2
    v4u g0 = {1u, lds_off, (unsigned)(ga & 0xffffffffu),
              (unsigned)(((ga >> 32) & 0x1ffffffu) | (2u << 30))};
    // D# group 1: data_size=2B; tensor_dim0=cols, tensor_dim1=rows,
    // tile_dim0=cols, tile_dim1=rows, tensor_dim0_stride=stride
    v8i g1 = {(int)(1u << 16),
              (int)((cols & 0xffffu) << 16),
              (int)(((cols >> 16) & 0xffffu) | ((rows & 0xffffu) << 16)),
              (int)(((rows >> 16) & 0xffffu) | ((cols & 0xffffu) << 16)),
              (int)(rows & 0xffffu),
              (int)stride,
              0, 0};
    v4i z4 = {0, 0, 0, 0};
#if __clang_major__ >= 23
    v8i z8 = {0, 0, 0, 0, 0, 0, 0, 0};
    __builtin_amdgcn_tensor_load_to_lds(g0, g1, z4, z4, z8, 0);
#else
    __builtin_amdgcn_tensor_load_to_lds(g0, g1, z4, z4, 0);
#endif
}
#endif

// ---------------- conversion kernels -------------------------------------
__global__ __launch_bounds__(256) void k_cvt_x(const float* __restrict__ x,
                                               _Float16* __restrict__ xt) {
    int idx = blockIdx.x * 256 + threadIdx.x;           // coalesced read
    int b = idx >> 20;
    int rem = idx & ((1 << 20) - 1);
    int c = rem >> 12;
    int n = rem & 4095;
    xt[((b << 12) + n) * 256 + c] = (_Float16)x[idx];
}

__global__ __launch_bounds__(256) void k_cvt(const float* __restrict__ src,
                                             _Float16* __restrict__ dst, int n) {
    int idx = blockIdx.x * 256 + threadIdx.x;
    if (idx < n) dst[idx] = (_Float16)src[idx];
}

// ---------------- K1: QKV GEMM (32 tokens/block) + focusing epilogue ------
__global__ __launch_bounds__(256) void k_qkv(const _Float16* __restrict__ xt,
                                             const _Float16* __restrict__ w16,
                                             const float* __restrict__ qkv_b,
                                             const float* __restrict__ scale,
                                             _Float16* __restrict__ qf,
                                             _Float16* __restrict__ kf,
                                             _Float16* __restrict__ vf) {
    __shared__ float s_scr[256];                         // 1/softplus(scale)
    __shared__ float s_bias[768];
    __shared__ float s_red[32][4];                       // q2,q6,k2,k6 per token
    __shared__ float s_fac[32][2];                       // sqrt(s1/s3) q,k
    __shared__ __align__(16) _Float16 s_A[32 * 32];
    __shared__ __align__(16) _Float16 s_B[768 * 32];

    int tid = threadIdx.x;
    int wv = tid >> 5, lane = tid & 31;

    {
        float s = scale[tid];
        s_scr[tid] = 1.0f / logf(1.0f + expf(s));
    }
    for (int i = tid; i < 768; i += 256) s_bias[i] = qkv_b[i];
    if (tid < 128) s_red[tid >> 2][tid & 3] = 0.0f;
    __syncthreads();

    int g0tok = blockIdx.x * 32;
    const _Float16* xrow = xt + (size_t)g0tok * 256;

    v8f acc[2][6];
#pragma unroll
    for (int mt = 0; mt < 2; ++mt)
#pragma unroll
        for (int j = 0; j < 6; ++j)
            acc[mt][j] = (v8f){0.f, 0.f, 0.f, 0.f, 0.f, 0.f, 0.f, 0.f};

    for (int k0 = 0; k0 < 256; k0 += 32) {
        if (tid < 128) {
            int r = tid >> 2, c8 = (tid & 3) * 8;
            *(uint4*)&s_A[r * 32 + c8] = *(const uint4*)&xrow[r * 256 + k0 + c8];
        }
#if USE_TDM
        if (wv == 0) {
            tdm_load_tile_f16((unsigned)(__SIZE_TYPE__)&s_B[0], w16 + k0, 768, 32, 256);
            __builtin_amdgcn_s_wait_tensorcnt(0);
        }
#else
        for (int idx = tid; idx < 768 * 4; idx += 256) {
            int o = idx >> 2, c8 = (idx & 3) * 8;
            *(uint4*)&s_B[o * 32 + c8] = *(const uint4*)&w16[o * 256 + k0 + c8];
        }
#endif
        __syncthreads();
        v16h a0 = load_A_frag(s_A, 32, lane);
        v16h a1 = load_A_frag(s_A + 16 * 32, 32, lane);
#pragma unroll
        for (int j = 0; j < 6; ++j) {
            v16h bf = load_B_frag_nk(s_B, 32, lane, wv * 96 + j * 16);
            acc[0][j] = __builtin_amdgcn_wmma_f32_16x16x32_f16(false, a0, false, bf,
                                                               (short)0, acc[0][j], false, false);
            acc[1][j] = __builtin_amdgcn_wmma_f32_16x16x32_f16(false, a1, false, bf,
                                                               (short)0, acc[1][j], false, false);
        }
        __syncthreads();
    }

    int rowbase = (lane < 16) ? 0 : 8;
    int col0 = lane & 15;

    // pass 1: per-token sum(u^2), sum(u^6), batched per thread then LDS atomics
#pragma unroll
    for (int mt = 0; mt < 2; ++mt)
#pragma unroll
        for (int r = 0; r < 8; ++r) {
            int m = mt * 16 + rowbase + r;
            float sq2 = 0.f, sq6 = 0.f, sk2 = 0.f, sk6 = 0.f;
#pragma unroll
            for (int j = 0; j < 6; ++j) {
                int o = wv * 96 + j * 16 + col0;
                if (o < 512) {
                    float val = acc[mt][j][r] + s_bias[o];
                    float u = (fmaxf(val, 0.0f) + EPSF) * s_scr[o & 255];
                    float u2 = u * u, u6 = u2 * u2 * u2;
                    if (o < 256) { sq2 += u2; sq6 += u6; }
                    else         { sk2 += u2; sk6 += u6; }
                }
            }
            if (sq2 > 0.f) { atomicAdd(&s_red[m][0], sq2); atomicAdd(&s_red[m][1], sq6); }
            if (sk2 > 0.f) { atomicAdd(&s_red[m][2], sk2); atomicAdd(&s_red[m][3], sk6); }
        }
    __syncthreads();
    if (tid < 64) {
        int m = tid >> 1, w = tid & 1;
        s_fac[m][w] = sqrtf(s_red[m][2 * w] / s_red[m][2 * w + 1]);
    }
    __syncthreads();

    // pass 2: finalize and write q,k,v head-major f16
#pragma unroll
    for (int mt = 0; mt < 2; ++mt)
#pragma unroll
        for (int j = 0; j < 6; ++j) {
            int o = wv * 96 + j * 16 + col0;
#pragma unroll
            for (int r = 0; r < 8; ++r) {
                int m = mt * 16 + rowbase + r;
                int g = g0tok + m;
                int bb = g >> 12;
                int n = g & 4095;
                float val = acc[mt][j][r] + s_bias[o];
                if (o < 512) {
                    int c = o & 255;
                    float u = (fmaxf(val, 0.0f) + EPSF) * s_scr[c];
                    float outv = u * u * u * s_fac[m][(o < 256) ? 0 : 1];
                    int head = c >> 5, d = c & 31;
                    size_t idx = ((size_t)(bb * 8 + head) * 4096 + n) * 32 + d;
                    if (o < 256) qf[idx] = (_Float16)outv;
                    else         kf[idx] = (_Float16)outv;
                } else {
                    int c = o - 512;
                    int head = c >> 5, d = c & 31;
                    vf[((size_t)(bb * 8 + head) * 4096 + n) * 32 + d] = (_Float16)val;
                }
            }
        }
}

// ---------------- K2: depthwise 5x5 conv on v ----------------------------
__global__ __launch_bounds__(256) void k_dwconv(const _Float16* __restrict__ vf,
                                                const float* __restrict__ dwc_w,
                                                const float* __restrict__ dwc_b,
                                                _Float16* __restrict__ fm) {
    __shared__ float w_s[32 * 25];
    __shared__ float b_s[32];
    int tid = threadIdx.x;
    for (int i = tid; i < 800; i += 256) w_s[i] = dwc_w[i];
    if (tid < 32) b_s[tid] = dwc_b[tid];
    __syncthreads();

    int idx = blockIdx.x * 256 + tid;                    // [bh][n][d]
    int d = idx & 31;
    int t = idx >> 5;
    int n = t & 4095;
    int bh = t >> 12;
    int i = n >> 6, jc = n & 63;
    const _Float16* vb = vf + ((size_t)bh << 12) * 32;
    float accv = b_s[d];
#pragma unroll
    for (int dy = 0; dy < 5; ++dy) {
        int yy = i + dy - 2;
        if (yy < 0 || yy >= 64) continue;
#pragma unroll
        for (int dx = 0; dx < 5; ++dx) {
            int xx = jc + dx - 2;
            if (xx < 0 || xx >= 64) continue;
            accv += (float)vb[(yy * 64 + xx) * 32 + d] * w_s[d * 25 + dy * 5 + dx];
        }
    }
    fm[idx] = (_Float16)accv;
}

// ---------------- K3: kv = k^T v and ksum, per (b,head) -------------------
__global__ __launch_bounds__(256) void k_kv(const _Float16* __restrict__ kf,
                                            const _Float16* __restrict__ vf,
                                            _Float16* __restrict__ kvb,
                                            float* __restrict__ ksum) {
    __shared__ __align__(16) unsigned char sbuf[32768];  // stage then kv-reduce
    __shared__ float s_ksum[32];
    int tid = threadIdx.x;
    int wv = tid >> 5, lane = tid & 31;
    int bh = blockIdx.x;
    const _Float16* kb_ = kf + ((size_t)bh << 12) * 32;
    const _Float16* vb_ = vf + ((size_t)bh << 12) * 32;

    _Float16* Ks = (_Float16*)sbuf + wv * 2048;          // [32][32]
    _Float16* Vs = Ks + 1024;                            // [32][32]

    v8f pkv[2][2];
#pragma unroll
    for (int ct = 0; ct < 2; ++ct)
#pragma unroll
        for (int dt = 0; dt < 2; ++dt)
            pkv[ct][dt] = (v8f){0.f, 0.f, 0.f, 0.f, 0.f, 0.f, 0.f, 0.f};

    int rowbase = (lane < 16) ? 0 : 8;
    int col0 = lane & 15;

    for (int jc = wv; jc < 128; jc += 8) {
        const uint4* sk = (const uint4*)&kb_[(jc * 32 + lane) * 32];
        const uint4* sv = (const uint4*)&vb_[(jc * 32 + lane) * 32];
        uint4* dk = (uint4*)&Ks[lane * 32];
        uint4* dv = (uint4*)&Vs[lane * 32];
#pragma unroll
        for (int p = 0; p < 4; ++p) { dk[p] = sk[p]; dv[p] = sv[p]; }
        __syncthreads();
#pragma unroll
        for (int ct = 0; ct < 2; ++ct) {
            v16h a;                                      // A[c,j] = Ks[j][c]
#pragma unroll
            for (int e = 0; e < 16; ++e)
                a[e] = Ks[a_kidx(e, lane) * 32 + ct * 16 + col0];
#pragma unroll
            for (int dt = 0; dt < 2; ++dt) {
                v16h b = load_B_frag_kn(Vs, 32, lane, dt * 16);
                pkv[ct][dt] = __builtin_amdgcn_wmma_f32_16x16x32_f16(
                    false, a, false, b, (short)0, pkv[ct][dt], false, false);
            }
        }
        __syncthreads();
    }

    float* kvred = (float*)sbuf;                         // reuse: 8 * 1024 f32
#pragma unroll
    for (int ct = 0; ct < 2; ++ct)
#pragma unroll
        for (int dt = 0; dt < 2; ++dt)
#pragma unroll
            for (int r = 0; r < 8; ++r) {
                int c = ct * 16 + rowbase + r;
                int d = dt * 16 + col0;
                kvred[wv * 1024 + c * 32 + d] = pkv[ct][dt][r];
            }
    if (tid < 32) s_ksum[tid] = 0.0f;
    __syncthreads();
    for (int i = tid; i < 1024; i += 256) {
        float s = 0.f;
#pragma unroll
        for (int w = 0; w < 8; ++w) s += kvred[w * 1024 + i];
        kvb[(size_t)bh * 1024 + i] = (_Float16)s;
    }
    {
        int c = tid & 31;
        int jbase = (tid >> 5) * 512;
        float partial = 0.f;
        for (int jj = 0; jj < 512; ++jj)
            partial += (float)kb_[(jbase + jj) * 32 + c];
        atomicAdd(&s_ksum[c], partial);
    }
    __syncthreads();
    if (tid < 32) ksum[bh * 32 + tid] = s_ksum[tid];
}

// ---------------- K4: out = (q @ kv) * z + fm, write [B,N,C] f16 ----------
__global__ __launch_bounds__(256) void k_att(const _Float16* __restrict__ qf,
                                             const _Float16* __restrict__ kvb,
                                             const float* __restrict__ ksum,
                                             const _Float16* __restrict__ fm,
                                             _Float16* __restrict__ att) {
    __shared__ __align__(16) _Float16 s_kv[1024];
    __shared__ float s_ks[32];
    __shared__ __align__(16) _Float16 s_q[8][16 * 32];

    int tid = threadIdx.x;
    int wv = tid >> 5, lane = tid & 31;
    int bh = blockIdx.x >> 5;
    int tch = blockIdx.x & 31;

    for (int i = tid; i < 1024; i += 256) s_kv[i] = kvb[(size_t)bh * 1024 + i];
    if (tid < 32) s_ks[tid] = ksum[bh * 32 + tid];

    int t0 = tch * 128 + wv * 16;
    const _Float16* qb_ = qf + ((size_t)bh * 4096 + t0) * 32;
    {
        int r = lane >> 1, hh = (lane & 1) * 16;
        uint4* dq = (uint4*)&s_q[wv][r * 32 + hh];
        const uint4* sq = (const uint4*)&qb_[r * 32 + hh];
        dq[0] = sq[0];
        dq[1] = sq[1];
    }
    __syncthreads();

    v16h a = load_A_frag(&s_q[wv][0], 32, lane);
    v8f o0 = (v8f){0.f, 0.f, 0.f, 0.f, 0.f, 0.f, 0.f, 0.f};
    v8f o1 = o0;
    {
        v16h b0 = load_B_frag_kn(s_kv, 32, lane, 0);
        v16h b1 = load_B_frag_kn(s_kv, 32, lane, 16);
        o0 = __builtin_amdgcn_wmma_f32_16x16x32_f16(false, a, false, b0, (short)0, o0, false, false);
        o1 = __builtin_amdgcn_wmma_f32_16x16x32_f16(false, a, false, b1, (short)0, o1, false, false);
    }
    float ps = 0.f;
#pragma unroll
    for (int e = 0; e < 16; ++e) ps += (float)a[e] * s_ks[a_kidx(e, lane)];
    ps += __shfl_xor(ps, 16, 32);
    float z = 1.0f / (ps + EPSF);

    int rowbase = (lane < 16) ? 0 : 8;
    int col0 = lane & 15;
    int b = bh >> 3, head = bh & 7;
#pragma unroll
    for (int dt = 0; dt < 2; ++dt) {
        v8f od = dt ? o1 : o0;
#pragma unroll
        for (int r = 0; r < 8; ++r) {
            int m = rowbase + r;
            int n = t0 + m;
            int d = dt * 16 + col0;
            float zz = __shfl(z, m, 32);
            float val = od[r] * zz + (float)fm[(((size_t)bh << 12) + n) * 32 + d];
            att[((size_t)b * 4096 + n) * 256 + head * 32 + d] = (_Float16)val;
        }
    }
}

// ---------------- K5: projection GEMM (32 tokens/block) -------------------
__global__ __launch_bounds__(256) void k_proj(const _Float16* __restrict__ att,
                                              const _Float16* __restrict__ wp16,
                                              const float* __restrict__ proj_b,
                                              float* __restrict__ out) {
    __shared__ float s_bias[256];
    __shared__ __align__(16) _Float16 s_A[32 * 32];
    __shared__ __align__(16) _Float16 s_B[256 * 32];

    int tid = threadIdx.x;
    int wv = tid >> 5, lane = tid & 31;
    s_bias[tid] = proj_b[tid];

    int g0tok = blockIdx.x * 32;
    const _Float16* arow = att + (size_t)g0tok * 256;

    v8f acc[2][2];
#pragma unroll
    for (int mt = 0; mt < 2; ++mt)
#pragma unroll
        for (int j = 0; j < 2; ++j)
            acc[mt][j] = (v8f){0.f, 0.f, 0.f, 0.f, 0.f, 0.f, 0.f, 0.f};

    for (int k0 = 0; k0 < 256; k0 += 32) {
        if (tid < 128) {
            int r = tid >> 2, c8 = (tid & 3) * 8;
            *(uint4*)&s_A[r * 32 + c8] = *(const uint4*)&arow[r * 256 + k0 + c8];
        }
#if USE_TDM
        if (wv == 0) {
            tdm_load_tile_f16((unsigned)(__SIZE_TYPE__)&s_B[0], wp16 + k0, 256, 32, 256);
            __builtin_amdgcn_s_wait_tensorcnt(0);
        }
#else
        for (int idx = tid; idx < 256 * 4; idx += 256) {
            int o = idx >> 2, c8 = (idx & 3) * 8;
            *(uint4*)&s_B[o * 32 + c8] = *(const uint4*)&wp16[o * 256 + k0 + c8];
        }
#endif
        __syncthreads();
        v16h a0 = load_A_frag(s_A, 32, lane);
        v16h a1 = load_A_frag(s_A + 16 * 32, 32, lane);
#pragma unroll
        for (int j = 0; j < 2; ++j) {
            v16h bf = load_B_frag_nk(s_B, 32, lane, wv * 32 + j * 16);
            acc[0][j] = __builtin_amdgcn_wmma_f32_16x16x32_f16(false, a0, false, bf,
                                                               (short)0, acc[0][j], false, false);
            acc[1][j] = __builtin_amdgcn_wmma_f32_16x16x32_f16(false, a1, false, bf,
                                                               (short)0, acc[1][j], false, false);
        }
        __syncthreads();
    }

    int rowbase = (lane < 16) ? 0 : 8;
    int col0 = lane & 15;
#pragma unroll
    for (int mt = 0; mt < 2; ++mt)
#pragma unroll
        for (int j = 0; j < 2; ++j) {
            int o = wv * 32 + j * 16 + col0;
#pragma unroll
            for (int r = 0; r < 8; ++r) {
                int g = g0tok + mt * 16 + rowbase + r;
                int bb = g >> 12;
                int n = g & 4095;
                out[((size_t)(bb * 256 + o)) * 4096 + n] = acc[mt][j][r] + s_bias[o];
            }
        }
}

// ---------------- host launch --------------------------------------------
extern "C" void kernel_launch(void* const* d_in, const int* in_sizes, int n_in,
                              void* d_out, int out_size, void* d_ws, size_t ws_size,
                              hipStream_t stream) {
    const float* x      = (const float*)d_in[0];
    const float* qkv_w  = (const float*)d_in[1];
    const float* qkv_b  = (const float*)d_in[2];
    const float* proj_w = (const float*)d_in[3];
    const float* proj_b = (const float*)d_in[4];
    const float* dwc_w  = (const float*)d_in[5];
    const float* dwc_b  = (const float*)d_in[6];
    const float* scale  = (const float*)d_in[7];
    float* outp = (float*)d_out;

    char* ws = (char*)d_ws;
    const size_t SZ_TOK = 16ull * 4096 * 256 * 2;
    const size_t SZ_HD  = 128ull * 4096 * 32 * 2;
    size_t off = 0;
    _Float16* xt   = (_Float16*)(ws + off); off += SZ_TOK;
    _Float16* w16  = (_Float16*)(ws + off); off += 768ull * 256 * 2;
    _Float16* wp16 = (_Float16*)(ws + off); off += 256ull * 256 * 2;
    _Float16* qf   = (_Float16*)(ws + off); off += SZ_HD;
    _Float16* kf   = (_Float16*)(ws + off); off += SZ_HD;
    _Float16* vf   = (_Float16*)(ws + off); off += SZ_HD;
    _Float16* fm   = (_Float16*)(ws + off); off += SZ_HD;
    _Float16* att  = (_Float16*)(ws + off); off += SZ_TOK;
    _Float16* kvb  = (_Float16*)(ws + off); off += 128ull * 1024 * 2;
    float*    ksum = (float*)(ws + off);    off += 128ull * 32 * 4;

    k_cvt_x<<<65536, 256, 0, stream>>>(x, xt);
    k_cvt<<<768, 256, 0, stream>>>(qkv_w, w16, 768 * 256);
    k_cvt<<<256, 256, 0, stream>>>(proj_w, wp16, 256 * 256);
    k_qkv<<<2048, 256, 0, stream>>>(xt, w16, qkv_b, scale, qf, kf, vf);
    k_dwconv<<<65536, 256, 0, stream>>>(vf, dwc_w, dwc_b, fm);
    k_kv<<<128, 256, 0, stream>>>(kf, vf, kvb, ksum);
    k_att<<<4096, 256, 0, stream>>>(qf, kvb, ksum, fm, att);
    k_proj<<<2048, 256, 0, stream>>>(att, wp16, proj_b, outp);
}
